// Decoder_30777735643309
// MI455X (gfx1250) — compile-verified, matrix operand order checked by reading, not run on hardware
//
#include <hip/hip_runtime.h>
#include <math.h>

// Problem constants (from the reference)
#define BB  128      // batch
#define TT  50       // timesteps
#define VV  10004    // vocab
#define EE  300      // embedding dim
#define EP  320      // embedding dim padded to 32 (K padding for WMMA)
#define SHH 1024     // session-encoding dim
#define HH  512      // hid_size
#define GG  1024     // GRU hidden (2*hid)
#define G3  3072     // 3*GG
#define VP  10048    // vocab padded to 64 (N-group padding, avoids OOB frag loads)

#define LDS_STRIDE 72          // 64 + 8 bf16: 144B row stride -> conflict-free ds_load_b128
#define CHUNK_ELEMS (64 * LDS_STRIDE)   // one 64x64 B chunk (padded) in LDS

typedef __attribute__((ext_vector_type(16))) __bf16 v16bf;
typedef __attribute__((ext_vector_type(8)))  float  v8f;

static __device__ __forceinline__ __bf16 f2bf(float f) {
  unsigned u = __builtin_bit_cast(unsigned, f);
  unsigned short s = (unsigned short)((u + 0x7FFFu + ((u >> 16) & 1u)) >> 16); // RNE
  return __builtin_bit_cast(__bf16, s);
}

// CDNA5 async memory->LDS copy (16B), tracked by ASYNCcnt. VDST = LDS byte
// address (low 32 bits of the flat pointer), VADDR = 64-bit global address.
static __device__ __forceinline__ void async_copy16(unsigned lds_off, const __bf16* g) {
  unsigned long long ga = (unsigned long long)(size_t)g;
  asm volatile("global_load_async_to_lds_b128 %0, %1, off"
               :: "v"(lds_off), "v"(ga) : "memory");
}

// Load one 16x32 bf16 WMMA operand fragment from a row-major [rows,K] matrix
// (global or LDS). CDNA5 16-bit operand layout:
//   lane L: row = L&15, K-half g = L>>4;
//   v[0..3] hold K = k0+8g .. k0+8g+7, v[4..7] hold K = k0+16+8g .. k0+16+8g+7.
// -> exactly two b128 loads per lane (all call sites are 16B aligned).
static __device__ __forceinline__ v16bf load_frag(const __bf16* M, int ld, int row0, int k0) {
  unsigned lane = threadIdx.x & 31u;
  int r = row0 + (int)(lane & 15u);
  int g = (int)(lane >> 4);
  const __bf16* p = M + (size_t)r * (size_t)ld + (size_t)(k0 + 8 * g);
  union { v16bf v; uint4 q[2]; } u;
  u.q[0] = *(const uint4*)(p);
  u.q[1] = *(const uint4*)(p + 16);
  return u.v;
}

static __device__ __forceinline__ v8f wmma_bf16(v16bf a, v16bf b, v8f c) {
  // D = A x B + C, f32 accumulate
  return __builtin_amdgcn_wmma_f32_16x16x32_bf16(false, a, false, b, (short)0, c, false, false);
}

// Issue the async copies for one 64(rows) x 64(K) chunk of W into an LDS
// buffer: 512 x 16B transfers = exactly 2 per thread (no loop / exec games).
static __device__ __forceinline__ void stage_chunk(const __bf16* W, int ldb, int n0, int k0,
                                                   __bf16* buf) {
  const int tid = threadIdx.x;
  const int r0 = tid >> 3, s0 = tid & 7;          // transfer tid
  async_copy16((unsigned)(size_t)(buf + r0 * LDS_STRIDE + s0 * 8),
               W + (size_t)(n0 + r0) * (size_t)ldb + (size_t)(k0 + s0 * 8));
  const int r1 = r0 + 32;                          // transfer tid+256
  async_copy16((unsigned)(size_t)(buf + r1 * LDS_STRIDE + s0 * 8),
               W + (size_t)(n0 + r1) * (size_t)ldb + (size_t)(k0 + s0 * 8));
}

// Consume one staged 64x64 chunk: load both A fragments (global) and all 8 B
// fragments (LDS) into registers first, then run the 8 WMMAs -- avoids a full
// s_wait_dscnt 0 in front of every WMMA.
static __device__ __forceinline__ void compute_chunk(const __bf16* A, int lda, int m0, int k0,
                                                     const __bf16* buf, v8f acc[4]) {
  const unsigned lane = threadIdx.x & 31u;
  const int g = (int)(lane >> 4), nlo = (int)(lane & 15u);
  v16bf a0 = load_frag(A, lda, m0, k0);
  v16bf a1 = load_frag(A, lda, m0, k0 + 32);
  v16bf b[2][4];
#pragma unroll
  for (int kt = 0; kt < 2; ++kt) {
#pragma unroll
    for (int c = 0; c < 4; ++c) {
      const __bf16* p = buf + (size_t)(16 * c + nlo) * LDS_STRIDE + (size_t)(kt * 32 + 8 * g);
      union { v16bf v; uint4 q[2]; } u;
      u.q[0] = *(const uint4*)(p);
      u.q[1] = *(const uint4*)(p + 16);
      b[kt][c] = u.v;
    }
  }
#pragma unroll
  for (int c = 0; c < 4; ++c) acc[c] = wmma_bf16(a0, b[0][c], acc[c]);
#pragma unroll
  for (int c = 0; c < 4; ++c) acc[c] = wmma_bf16(a1, b[1][c], acc[c]);
}

// Block-cooperative GEMM core: 256 threads = 8 waves; wave w computes the
// 16x64 tile (m0 = 16*w, n columns n0..n0+63) of A[128,K] @ W[N,K]^T.
// The 64x64 bf16 chunks of W (shared by all 8 waves) are DOUBLE-BUFFERED in
// LDS via async global->LDS b128 copies: chunk i+1 streams in (ASYNCcnt) while
// chunk i feeds the WMMAs from LDS. In-order async completion lets us wait
// ASYNCcnt<=2 (the 2 outstanding copies of the next chunk) instead of 0.
static __device__ __forceinline__ void gemm_lds(const __bf16* A, int lda, int m0,
                                                const __bf16* W, int ldb, int n0,
                                                int K, __bf16* sB, v8f acc[4]) {
  const int nchunks = K >> 6;
  stage_chunk(W, ldb, n0, 0, sB);
  for (int i = 0; i < nchunks; ++i) {
    __bf16* cur = sB + (size_t)(i & 1) * CHUNK_ELEMS;
    __bf16* nxt = sB + (size_t)((i + 1) & 1) * CHUNK_ELEMS;
    if (i + 1 < nchunks) {
      stage_chunk(W, ldb, n0, (i + 1) << 6, nxt);
      asm volatile("s_wait_asynccnt 0x2" ::: "memory");  // chunk i landed; i+1 in flight
    } else {
      asm volatile("s_wait_asynccnt 0x0" ::: "memory");
    }
    __syncthreads();                                     // all waves' copies visible
    compute_chunk(A, lda, m0, i << 6, cur, acc);
    __syncthreads();                                     // seal reads before buffer reuse
  }
}

// ---------------- conversion / gather kernels ----------------

__global__ void cvt_pad_kernel(__bf16* dst, const float* src,
                               int srcR, int srcC, int dstR, int dstC) {
  int idx = blockIdx.x * blockDim.x + threadIdx.x;
  int n = dstR * dstC;
  if (idx >= n) return;
  int r = idx / dstC, c = idx % dstC;
  float v = (r < srcR && c < srcC) ? src[(size_t)r * srcC + c] : 0.0f;
  dst[idx] = f2bf(v);
}

__global__ void gather_emb_kernel(__bf16* e_b, const float* emb, const int* x) {
  int idx = blockIdx.x * blockDim.x + threadIdx.x;
  if (idx >= BB * TT * EP) return;
  int c  = idx % EP;
  int bt = idx / EP;
  int tok = x[bt];
  float v = (c < EE) ? emb[(size_t)tok * EE + c] : 0.0f;
  e_b[idx] = f2bf(v);
}

// ---------------- ses = tanh(senc @ W1^T + b1); Hfull = [ses|ses] ----------------

__global__ void ses_kernel(const __bf16* senc_b, const __bf16* W1_b, const float* b1,
                           float* Hfull_f, __bf16* Hfull_b) {
  __shared__ __align__(16) __bf16 sB[2 * CHUNK_ELEMS];
  int w  = threadIdx.x >> 5;      // 8 waves per block -> 8 M tiles (covers all 128 rows)
  int m0 = w * 16;
  int n0 = blockIdx.x * 64;       // 16x64 tile per wave
  v8f acc[4] = {};
  gemm_lds(senc_b, SHH, m0, W1_b, SHH, n0, SHH, sB, acc);
  unsigned lane = threadIdx.x & 31u;
  int nlo = (int)(lane & 15u), g = (int)(lane >> 4);
  for (int c = 0; c < 4; ++c) {
    int n = n0 + 16 * c + nlo;
    float bias = b1[n];
    for (int r = 0; r < 8; ++r) {
      int m = m0 + r + 8 * g;
      float v = tanhf(acc[c][r] + bias);
      __bf16 vb = f2bf(v);
      Hfull_f[(size_t)m * GG + n]      = v;   // h0 = ses
      Hfull_f[(size_t)m * GG + HH + n] = v;   // upper half = ses (constant forever)
      Hfull_b[(size_t)m * GG + n]      = vb;
      Hfull_b[(size_t)m * GG + HH + n] = vb;
    }
  }
}

// ---------------- GH = Hfull@Whh^T, GX = e_t@Wih^T (one kernel) ----------------

__global__ void gxgh_kernel(const __bf16* Hfull_b, const __bf16* Whh_b,
                            const __bf16* e_b, const __bf16* Wih_b,
                            float* GH, float* GX, int t) {
  __shared__ __align__(16) __bf16 sB[2 * CHUNK_ELEMS];
  int w  = threadIdx.x >> 5;
  int m0 = w * 16;
  int n0 = blockIdx.x * 64;                 // 48 blocks cover N=3072
  v8f accH[4] = {};
  v8f accX[4] = {};
  // recurrent path, K=1024
  gemm_lds(Hfull_b, GG, m0, Whh_b, GG, n0, GG, sB, accH);
  // input path, K=320 (padded 300); batch-row stride of e is TT*EP
  const __bf16* Et = e_b + (size_t)t * EP;
  gemm_lds(Et, TT * EP, m0, Wih_b, EP, n0, EP, sB, accX);
  unsigned lane = threadIdx.x & 31u;
  int nlo = (int)(lane & 15u), g = (int)(lane >> 4);
  for (int c = 0; c < 4; ++c) {
    int n = n0 + 16 * c + nlo;
    for (int r = 0; r < 8; ++r) {
      int m = m0 + r + 8 * g;
      GH[(size_t)m * G3 + n] = accH[c][r];
      GX[(size_t)m * G3 + n] = accX[c][r];
    }
  }
}

// ---------------- GRU gate math + state update ----------------

__global__ void gate_kernel(const float* GX, const float* GH,
                            const float* bih, const float* bhh,
                            float* Hfull_f, __bf16* Hfull_b, __bf16* hnew_b) {
  int idx = blockIdx.x * blockDim.x + threadIdx.x;
  if (idx >= BB * GG) return;
  int m = idx >> 10, j = idx & (GG - 1);
  size_t base = (size_t)m * G3;
  float gr  = GX[base + j]           + bih[j]           + GH[base + j]           + bhh[j];
  float gz  = GX[base + GG + j]      + bih[GG + j]      + GH[base + GG + j]      + bhh[GG + j];
  float gxn = GX[base + 2 * GG + j]  + bih[2 * GG + j];
  float ghn = GH[base + 2 * GG + j]  + bhh[2 * GG + j];
  float r = 1.0f / (1.0f + expf(-gr));
  float z = 1.0f / (1.0f + expf(-gz));
  float nv = tanhf(gxn + r * ghn);
  float hprev = Hfull_f[(size_t)m * GG + j];
  float hnew  = (1.0f - z) * nv + z * hprev;
  hnew_b[(size_t)m * GG + j] = f2bf(hnew);
  if (j < HH) {  // only first half carries to next step; upper half stays = ses
    Hfull_f[(size_t)m * GG + j] = hnew;
    Hfull_b[(size_t)m * GG + j] = f2bf(hnew);
  }
}

// ---------------- o = hnew@W2^T + b2 + e_t  (bf16 out, padded cols zeroed) ----------------

__global__ void o_kernel(const __bf16* hnew_b, const __bf16* W2_b, const float* b2,
                         const float* emb, const int* x, __bf16* o_b, int t) {
  __shared__ __align__(16) __bf16 sB[2 * CHUNK_ELEMS];
  int w  = threadIdx.x >> 5;
  int m0 = w * 16;
  int n0 = blockIdx.x * 64;                 // 5 blocks cover N=320
  v8f acc[4] = {};
  gemm_lds(hnew_b, GG, m0, W2_b, GG, n0, GG, sB, acc);
  unsigned lane = threadIdx.x & 31u;
  int nlo = (int)(lane & 15u), g = (int)(lane >> 4);
  for (int c = 0; c < 4; ++c) {
    int n = n0 + 16 * c + nlo;
    for (int r = 0; r < 8; ++r) {
      int m = m0 + r + 8 * g;
      float v = 0.0f;
      if (n < EE) {
        int tok = x[m * TT + t];
        v = acc[c][r] + b2[n] + emb[(size_t)tok * EE + n];  // bias + residual emb
      }
      o_b[(size_t)m * EP + n] = f2bf(v);    // cols 300..319 exactly zero
    }
  }
}

// ---------------- logits = o @ Wout^T -> d_out[b,t,:] ----------------

__global__ void logits_kernel(const __bf16* o_b, const __bf16* Wout_b, float* out, int t) {
  __shared__ __align__(16) __bf16 sB[2 * CHUNK_ELEMS];
  int w  = threadIdx.x >> 5;
  int m0 = w * 16;
  int n0 = blockIdx.x * 64;                 // 157 blocks cover VP=10048
  v8f acc[4] = {};
  gemm_lds(o_b, EP, m0, Wout_b, EP, n0, EP, sB, acc);
  unsigned lane = threadIdx.x & 31u;
  int nlo = (int)(lane & 15u), g = (int)(lane >> 4);
  for (int c = 0; c < 4; ++c) {
    int n = n0 + 16 * c + nlo;
    if (n < VV) {
      for (int r = 0; r < 8; ++r) {
        int m = m0 + r + 8 * g;
        out[((size_t)m * TT + t) * VV + n] = acc[c][r];  // 16-lane contiguous stores
      }
    }
  }
}

// ---------------- host side ----------------

extern "C" void kernel_launch(void* const* d_in, const int* in_sizes, int n_in,
                              void* d_out, int out_size, void* d_ws, size_t ws_size,
                              hipStream_t stream) {
  (void)in_sizes; (void)n_in; (void)out_size; (void)ws_size;
  const float* senc = (const float*)d_in[0];   // [1,128,1024]
  const int*   x    = (const int*)  d_in[1];   // [128,50]
  const float* emb  = (const float*)d_in[2];   // [10004,300]
  const float* Wih  = (const float*)d_in[3];   // [3072,300]
  const float* Whh  = (const float*)d_in[4];   // [3072,1024]
  const float* bih  = (const float*)d_in[5];   // [3072]
  const float* bhh  = (const float*)d_in[6];   // [3072]
  const float* W1   = (const float*)d_in[7];   // [512,1024]
  const float* b1   = (const float*)d_in[8];   // [512]
  const float* W2   = (const float*)d_in[9];   // [300,1024]
  const float* b2   = (const float*)d_in[10];  // [300]
  const float* Wout = (const float*)d_in[11];  // [10004,300]
  float* out = (float*)d_out;                  // [128,50,10004]

  char* base = (char*)d_ws;
  size_t off = 0;
  auto alloc = [&](size_t elems, size_t esz) -> char* {
    char* p = base + off;
    off = (off + elems * esz + 255) & ~(size_t)255;
    return p;
  };
  __bf16* Wih_b   = (__bf16*)alloc((size_t)G3 * EP, 2);
  __bf16* Whh_b   = (__bf16*)alloc((size_t)G3 * GG, 2);
  __bf16* W2_b    = (__bf16*)alloc((size_t)EP * GG, 2);
  __bf16* Wout_b  = (__bf16*)alloc((size_t)VP * EP, 2);
  __bf16* W1_b    = (__bf16*)alloc((size_t)HH * SHH, 2);
  __bf16* senc_b  = (__bf16*)alloc((size_t)BB * SHH, 2);
  __bf16* e_b     = (__bf16*)alloc((size_t)BB * TT * EP, 2);
  float*  Hfull_f = (float*) alloc((size_t)BB * GG, 4);
  __bf16* Hfull_b = (__bf16*)alloc((size_t)BB * GG, 2);
  __bf16* hnew_b  = (__bf16*)alloc((size_t)BB * GG, 2);
  float*  GH      = (float*) alloc((size_t)BB * G3, 4);
  float*  GX      = (float*) alloc((size_t)BB * G3, 4);
  __bf16* o_b     = (__bf16*)alloc((size_t)BB * EP, 2);

  const int TPB = 256;
  auto blocks = [](size_t n) { return (unsigned)((n + 255) / 256); };

  // Weight / activation conversion to padded bf16 (one-time per call; L2-resident after)
  cvt_pad_kernel<<<blocks((size_t)G3 * EP), TPB, 0, stream>>>(Wih_b,  Wih,  G3,  EE,  G3, EP);
  cvt_pad_kernel<<<blocks((size_t)G3 * GG), TPB, 0, stream>>>(Whh_b,  Whh,  G3,  GG,  G3, GG);
  cvt_pad_kernel<<<blocks((size_t)EP * GG), TPB, 0, stream>>>(W2_b,   W2,   EE,  GG,  EP, GG);
  cvt_pad_kernel<<<blocks((size_t)VP * EP), TPB, 0, stream>>>(Wout_b, Wout, VV,  EE,  VP, EP);
  cvt_pad_kernel<<<blocks((size_t)HH * SHH), TPB, 0, stream>>>(W1_b,  W1,   HH,  SHH, HH, SHH);
  cvt_pad_kernel<<<blocks((size_t)BB * SHH), TPB, 0, stream>>>(senc_b, senc, BB, SHH, BB, SHH);
  gather_emb_kernel<<<blocks((size_t)BB * TT * EP), TPB, 0, stream>>>(e_b, emb, x);

  // ses + initial state
  ses_kernel<<<HH / 64, TPB, 0, stream>>>(senc_b, W1_b, b1, Hfull_f, Hfull_b);

  // Sequential recurrence
  for (int t = 0; t < TT; ++t) {
    gxgh_kernel<<<G3 / 64, TPB, 0, stream>>>(Hfull_b, Whh_b, e_b, Wih_b, GH, GX, t);
    gate_kernel<<<blocks((size_t)BB * GG), TPB, 0, stream>>>(GX, GH, bih, bhh,
                                                             Hfull_f, Hfull_b, hnew_b);
    o_kernel<<<EP / 64, TPB, 0, stream>>>(hnew_b, W2_b, b2, emb, x, o_b, t);
    logits_kernel<<<VP / 64, TPB, 0, stream>>>(o_b, Wout_b, out, t);
  }
}
